// MambaConvBlock_87789131531047
// MI455X (gfx1250) — matrix operand order repaired and checked by reading, not compiled, not run on hardware
//
#include <hip/hip_runtime.h>
#include <hip/hip_bf16.h>

// ---------------- shapes ----------------
#define Bsz   8
#define CIN   64
#define COUT  128
#define Hh    64
#define Ww    64
#define Lpix  4096      // Hh*Ww
#define DIc   256
#define Ndim  16
#define Rdim  8
#define Kdir  4
#define EPSv  1e-5f
#define BSTR  40        // LDS B-tile row stride (elements), padded: 80B, 16B aligned

typedef __attribute__((ext_vector_type(16))) __bf16 bf16x16;
typedef __attribute__((ext_vector_type(8)))  __bf16 bf16x8;
typedef __attribute__((ext_vector_type(8)))  float  floatx8;

__device__ __forceinline__ floatx8 wmma_bf16(bf16x16 a, bf16x16 b, floatx8 c) {
  return __builtin_amdgcn_wmma_f32_16x16x32_bf16(false, a, false, b, (short)0, c, false, false);
}

// A-matrix 16x32 bf16 fragment (row-major source, row stride lda):
// lane m=lane&15 holds row m; K(i) = (i<8?0:16) + 8*(lane>>4) + (i&7)
__device__ __forceinline__ bf16x16 load_a_frag(const __bf16* __restrict__ A,
                                               int lda, int row0, int k0,
                                               unsigned lane) {
  int m  = row0 + (int)(lane & 15u);
  int kb = (int)((lane >> 4) & 1u) * 8;
  const __bf16* p = A + (size_t)m * lda + k0 + kb;
  bf16x16 a;
#pragma unroll
  for (int i = 0; i < 8; ++i) a[i] = p[i];
#pragma unroll
  for (int i = 0; i < 8; ++i) a[8 + i] = p[16 + i];
  return a;
}

// B fragment from LDS tile stored K-major: Bs[n*BSTR + k], 32x16 tile.
// lane n=lane&15 needs K(i) = (i<8?0:16) + 8*(lane>>4) + (i&7) -> 2x ds_load_b128.
__device__ __forceinline__ bf16x16 load_b_frag_lds(const __bf16* Bs, unsigned lane) {
  const __bf16* bp = Bs + (int)(lane & 15u) * BSTR + (int)(lane >> 4) * 8;
  bf16x8 lo = *(const bf16x8*)bp;
  bf16x8 hi = *(const bf16x8*)(bp + 16);
  return __builtin_shufflevector(lo, hi, 0, 1, 2, 3, 4, 5, 6, 7,
                                 8, 9, 10, 11, 12, 13, 14, 15);
}

// ---------------------------------------------------------------
__global__ void cast_to_bf16(const float* __restrict__ in, __bf16* __restrict__ out, int n) {
  int i = blockIdx.x * blockDim.x + threadIdx.x;
  if (i < n) out[i] = (__bf16)in[i];
}

// conv weight restage: in [co][ci][tap] (tap=dy*3+dx) -> out [co][tap][ci]
__global__ void permute_conv_w_bf16(const float* __restrict__ in, __bf16* __restrict__ out,
                                    int Cout_, int Cin_) {
  int i = blockIdx.x * blockDim.x + threadIdx.x;
  int tot = Cout_ * Cin_ * 9;
  if (i >= tot) return;
  int co  = i / (Cin_ * 9);
  int r   = i - co * (Cin_ * 9);
  int tap = r / Cin_;
  int ci  = r - tap * Cin_;
  out[i] = (__bf16)in[(co * Cin_ + ci) * 9 + tap];
}

__global__ void zero_f32(float* __restrict__ p, int n) {
  int i = blockIdx.x * blockDim.x + threadIdx.x;
  if (i < n) p[i] = 0.f;
}

// ---------------------------------------------------------------
// conv 3x3 (SAME) + BN + ReLU, implicit GEMM, K ordered (tap, ci).
// Block = 4 waves; waves cover Cout (4 x 32 rows), share one 32x16 B tile
// staged K-major in LDS (double-buffered), halo masked at staging.
// ---------------------------------------------------------------
__global__ void conv_bn_relu_wmma(const __bf16* __restrict__ xin,
                                  const __bf16* __restrict__ wgt,
                                  const float* __restrict__ g,  const float* __restrict__ bta,
                                  const float* __restrict__ mn, const float* __restrict__ vr,
                                  float* __restrict__ out, int Cin) {
  __shared__ __align__(16) __bf16 Bs[2][16 * BSTR];
  const int Kdim = Cin * 9;
  const int cpt  = Cin >> 5;          // 32-chunks per tap
  const int nch  = cpt * 9;
  unsigned lane = threadIdx.x;
  int tid  = (int)(threadIdx.y * 32 + lane);
  int co0  = (int)threadIdx.y * 32;
  int t    = blockIdx.x;
  int b    = t >> 8;
  int pix0 = (t & 255) * 16;
  int h  = pix0 >> 6;
  int w0 = pix0 & 63;
  int n  = (int)(lane & 15u);

  auto stage = [&](int kc, int buf) {
    int tap = kc / cpt;
    int c0  = (kc - tap * cpt) << 5;
    int dy = tap / 3 - 1, dx = tap % 3 - 1;
    int hh = h + dy;
    bool rowok = (hh >= 0) && (hh < Hh);
    int hhc = min(max(hh, 0), Hh - 1);
    const __bf16* src = xin + (((size_t)(b * Cin + c0)) << 12) + (hhc << 6);
#pragma unroll
    for (int q = tid; q < 512; q += 128) {
      int kk = q >> 4, nn = q & 15;
      int ww = w0 + nn + dx;
      bool ok = rowok && (ww >= 0) && (ww < Ww);
      int wwc = min(max(ww, 0), Ww - 1);
      __bf16 v = src[((size_t)kk << 12) + wwc];
      Bs[buf][nn * BSTR + kk] = ok ? v : (__bf16)0.0f;
    }
  };

  floatx8 acc0 = {}, acc1 = {};
  stage(0, 0);
  for (int kc = 0; kc < nch; ++kc) {
    __syncthreads();
    if (kc + 1 < nch) stage(kc + 1, (kc + 1) & 1);
    int k0 = kc * 32;
    bf16x16 a0 = load_a_frag(wgt, Kdim, co0,      k0, lane);
    bf16x16 a1 = load_a_frag(wgt, Kdim, co0 + 16, k0, lane);
    bf16x16 bf = load_b_frag_lds(&Bs[kc & 1][0], lane);
    acc0 = wmma_bf16(a0, bf, acc0);
    acc1 = wmma_bf16(a1, bf, acc1);
  }
#pragma unroll
  for (int r = 0; r < 8; ++r) {
    int co  = co0 + r + 8 * (int)(lane >> 4);
    float sc = g[co] * rsqrtf(vr[co] + EPSv);
    float v  = acc0[r] * sc + (bta[co] - mn[co] * sc);
    out[(((size_t)b * COUT + co) << 12) + pix0 + n] = fmaxf(v, 0.f);
    int co2 = co + 16;
    float sc2 = g[co2] * rsqrtf(vr[co2] + EPSv);
    float v2  = acc1[r] * sc2 + (bta[co2] - mn[co2] * sc2);
    out[(((size_t)b * COUT + co2) << 12) + pix0 + n] = fmaxf(v2, 0.f);
  }
}

// ---------------------------------------------------------------
// channels-first LayerNorm over C channels at each (b, pixel) -> bf16
// ---------------------------------------------------------------
__global__ void ln_channels_bf16(const float* __restrict__ xin,
                                 const float* __restrict__ g, const float* __restrict__ bta,
                                 __bf16* __restrict__ out, int C) {
  int idx = blockIdx.x * blockDim.x + threadIdx.x;
  if (idx >= Bsz * Lpix) return;
  int b = idx >> 12, l = idx & 4095;
  const float* p = xin + (((size_t)b * C) << 12) + l;
  float mu = 0.f;
  for (int c = 0; c < C; ++c) mu += p[(size_t)c << 12];
  mu /= (float)C;
  float var = 0.f;
  for (int c = 0; c < C; ++c) { float d = p[(size_t)c << 12] - mu; var += d * d; }
  var /= (float)C;
  float inv = rsqrtf(var + EPSv);
  for (int c = 0; c < C; ++c)
    out[(((size_t)(b * C + c)) << 12) + l] =
        (__bf16)((p[(size_t)c << 12] - mu) * inv * g[c] + bta[c]);
}

// ---------------------------------------------------------------
// dense GEMM: out[b,m,l] = sum_k W[m,k]*X[b,k,l] + bias[m] (+resid)
// Block = 4 waves; B tile staged K-major to LDS, double-buffered.
// ---------------------------------------------------------------
__global__ void gemm_rowL_wmma(const __bf16* __restrict__ W, const __bf16* __restrict__ X,
                               const float* __restrict__ bias, const float* __restrict__ resid,
                               float* __restrict__ out, int M, int Kdim) {
  __shared__ __align__(16) __bf16 Bs[2][16 * BSTR];
  unsigned lane = threadIdx.x;
  int tid = (int)(threadIdx.y * 32 + lane);
  int m0 = (blockIdx.y * blockDim.y + threadIdx.y) * 32;
  int t  = blockIdx.x;
  int b  = t >> 8;
  int l0 = (t & 255) * 16;
  int n  = (int)(lane & 15u);
  const __bf16* Xbase = X + (((size_t)b * Kdim) << 12) + l0;
  const int nch = Kdim >> 5;

  auto stage = [&](int kc, int buf) {
#pragma unroll
    for (int q = tid; q < 256; q += 128) {
      int k = q >> 3, dw = q & 7;
      union { uint32_t u; __bf16 h[2]; } cv;
      cv.u = *((const uint32_t*)(Xbase + (((size_t)(kc * 32 + k)) << 12)) + dw);
      Bs[buf][(2 * dw)     * BSTR + k] = cv.h[0];
      Bs[buf][(2 * dw + 1) * BSTR + k] = cv.h[1];
    }
  };

  floatx8 acc0 = {}, acc1 = {};
  stage(0, 0);
  for (int kc = 0; kc < nch; ++kc) {
    __syncthreads();
    if (kc + 1 < nch) stage(kc + 1, (kc + 1) & 1);
    int k0 = kc * 32;
    bf16x16 a0 = load_a_frag(W, Kdim, m0,      k0, lane);
    bf16x16 a1 = load_a_frag(W, Kdim, m0 + 16, k0, lane);
    bf16x16 bf = load_b_frag_lds(&Bs[kc & 1][0], lane);
    acc0 = wmma_bf16(a0, bf, acc0);
    acc1 = wmma_bf16(a1, bf, acc1);
  }
#pragma unroll
  for (int r = 0; r < 8; ++r) {
    int m = m0 + r + 8 * (int)(lane >> 4);
    size_t o = (((size_t)b * M + m) << 12) + l0 + n;
    float v = acc0[r] + bias[m];
    if (resid) v += resid[o];
    out[o] = v;
    int m2 = m + 16;
    size_t o2 = (((size_t)b * M + m2) << 12) + l0 + n;
    float v2 = acc1[r] + bias[m2];
    if (resid) v2 += resid[o2];
    out[o2] = v2;
  }
}

// ---------------------------------------------------------------
// depthwise 3x3 conv on xg (= xz[:, :DI]) + bias + SiLU -> xc f32 + bf16
// ---------------------------------------------------------------
__global__ void dwconv_silu(const float* __restrict__ xz, const float* __restrict__ dww,
                            const float* __restrict__ dwb, float* __restrict__ xc,
                            __bf16* __restrict__ xcb) {
  int idx = blockIdx.x * blockDim.x + threadIdx.x;
  if (idx >= Bsz * DIc * Lpix) return;
  int l = idx & 4095;
  int bd = idx >> 12;
  int d = bd & 255;
  int b = bd >> 8;
  int h = l >> 6, w = l & 63;
  const float* src = xz + (((size_t)(b * 2 * DIc + d)) << 12);
  float acc = dwb[d];
#pragma unroll
  for (int dy = -1; dy <= 1; ++dy)
#pragma unroll
    for (int dx = -1; dx <= 1; ++dx) {
      int hh = h + dy, ww = w + dx;
      bool ok = (hh >= 0) && (hh < Hh) && (ww >= 0) && (ww < Ww);
      int hhc = min(max(hh, 0), Hh - 1);
      int wwc = min(max(ww, 0), Ww - 1);
      float v = src[(hhc << 6) + wwc];
      acc += ok ? dww[d * 9 + (dy + 1) * 3 + (dx + 1)] * v : 0.f;
    }
  float v = acc / (1.f + expf(-acc));    // SiLU
  xc[(size_t)idx]  = v;
  xcb[(size_t)idx] = (__bf16)v;
}

// ---------------------------------------------------------------
// x_proj: x_dbl[b,k,c,l] = sum_d Wxp[k,c,d] * xs[b,k,d,l]
// 3 waves (16-row M tiles over 40 rows) share one direction-gathered B tile in LDS.
// ---------------------------------------------------------------
__global__ void xproj_wmma(const __bf16* __restrict__ Wxp, const __bf16* __restrict__ xcb,
                           float* __restrict__ xdbl) {
  __shared__ __align__(16) __bf16 Bs[2][16 * BSTR];
  unsigned lane = threadIdx.x;
  int tid = (int)(threadIdx.y * 32 + lane);   // 0..95
  int m0 = (int)threadIdx.y * 16;
  int t  = blockIdx.x;
  int ltile = t & 255;
  int bk = t >> 8;
  int k  = bk & 3;
  int b  = bk >> 2;
  int l0 = ltile * 16;
  int n  = (int)(lane & 15u);
  const __bf16* src = xcb + (((size_t)b * DIc) << 12);
  const __bf16* Wk  = Wxp + (size_t)k * 40 * DIc;

  auto stage = [&](int kc, int buf) {
    for (int q = tid; q < 512; q += 96) {
      int kk = q >> 4, nn = q & 15;
      int l   = l0 + nn;
      int lt  = ((l & 63) << 6) | (l >> 6);
      int lf  = 4095 - l;
      int lft = ((lf & 63) << 6) | (lf >> 6);
      int idx = (k == 0) ? l : (k == 1) ? lt : (k == 2) ? lf : lft;
      Bs[buf][nn * BSTR + kk] = src[((size_t)(kc * 32 + kk) << 12) + idx];
    }
  };

  int m  = m0 + (int)(lane & 15u);
  bool mok = (m < 40);
  int mc = min(m, 39);
  const __bf16* pa0 = Wk + (size_t)mc * DIc + (int)((lane >> 4) & 1u) * 8;

  floatx8 acc = {};
  stage(0, 0);
  for (int kc = 0; kc < (DIc >> 5); ++kc) {
    __syncthreads();
    if (kc + 1 < (DIc >> 5)) stage(kc + 1, (kc + 1) & 1);
    bf16x16 a;
    const __bf16* pa = pa0 + kc * 32;
#pragma unroll
    for (int i = 0; i < 8; ++i) { __bf16 v = pa[i];      a[i]     = mok ? v : (__bf16)0.0f; }
#pragma unroll
    for (int i = 0; i < 8; ++i) { __bf16 v = pa[16 + i]; a[8 + i] = mok ? v : (__bf16)0.0f; }
    bf16x16 bf = load_b_frag_lds(&Bs[kc & 1][0], lane);
    acc = wmma_bf16(a, bf, acc);
  }
#pragma unroll
  for (int r = 0; r < 8; ++r) {
    int c = m0 + r + 8 * (int)(lane >> 4);
    if (c < 40)
      xdbl[(((size_t)bk * 40 + c) << 12) + l0 + n] = acc[r];
  }
}

// ---------------------------------------------------------------
// selective scan: one thread per (b,k,d); N=16 states in VGPRs.
// ---------------------------------------------------------------
__global__ void selective_scan(const float* __restrict__ xdbl, const float* __restrict__ xc,
                               const float* __restrict__ dtw, const float* __restrict__ dtb,
                               const float* __restrict__ A_log, const float* __restrict__ Ds,
                               float* __restrict__ y) {
  int idx = blockIdx.x * blockDim.x + threadIdx.x;
  if (idx >= Bsz * Kdir * DIc) return;
  int d  = idx & 255;
  int bk = idx >> 8;
  int k  = bk & 3;
  int b  = bk >> 2;

  float wdt[Rdim];
#pragma unroll
  for (int r = 0; r < Rdim; ++r) wdt[r] = dtw[(k * DIc + d) * Rdim + r];
  float dbias = dtb[k * DIc + d];
  float Arow[Ndim], hst[Ndim];
#pragma unroll
  for (int nn = 0; nn < Ndim; ++nn) {
    Arow[nn] = -expf(A_log[(k * DIc + d) * Ndim + nn]);
    hst[nn]  = 0.f;
  }
  float dsv = Ds[k * DIc + d];

  const float* xd     = xdbl + (((size_t)bk * 40) << 12);
  const float* u_base = xc + (((size_t)(b * DIc + d)) << 12);
  float*       yb     = y  + (((size_t)(b * DIc + d)) << 12);

  for (int l = 0; l < Lpix; ++l) {
    const float* sp = xd + l;
    float s = dbias;
#pragma unroll
    for (int r = 0; r < Rdim; ++r) s += wdt[r] * sp[(size_t)r << 12];
    float dt = (s > 20.f) ? s : logf(1.f + expf(s));   // softplus

    int lt  = ((l & 63) << 6) | (l >> 6);
    int lf  = 4095 - l;
    int lft = ((lf & 63) << 6) | (lf >> 6);
    int io  = (k == 0) ? l : (k == 1) ? lt : (k == 2) ? lf : lft;

    float u  = u_base[io];
    float du = dt * u;
    float yv = dsv * u;
#pragma unroll
    for (int nn = 0; nn < Ndim; ++nn) {
      float Bv = sp[(size_t)(Rdim + nn) << 12];
      float Cv = sp[(size_t)(Rdim + Ndim + nn) << 12];
      hst[nn] = expf(dt * Arow[nn]) * hst[nn] + du * Bv;
      yv += hst[nn] * Cv;
    }
    atomicAdd(yb + io, yv);
  }
}

// ---------------------------------------------------------------
// out_norm LN over DI channels * SiLU(z) -> bf16
// ---------------------------------------------------------------
__global__ void ln_gate_bf16(const float* __restrict__ y, const float* __restrict__ xz,
                             const float* __restrict__ g, const float* __restrict__ bta,
                             __bf16* __restrict__ out) {
  int idx = blockIdx.x * blockDim.x + threadIdx.x;
  if (idx >= Bsz * Lpix) return;
  int b = idx >> 12, l = idx & 4095;
  const float* p  = y  + (((size_t)b * DIc) << 12) + l;
  const float* zp = xz + (((size_t)(b * 2 * DIc + DIc)) << 12) + l;
  float mu = 0.f;
  for (int c = 0; c < DIc; ++c) mu += p[(size_t)c << 12];
  mu /= (float)DIc;
  float var = 0.f;
  for (int c = 0; c < DIc; ++c) { float d = p[(size_t)c << 12] - mu; var += d * d; }
  var /= (float)DIc;
  float inv = rsqrtf(var + EPSv);
  for (int c = 0; c < DIc; ++c) {
    float v = (p[(size_t)c << 12] - mu) * inv * g[c] + bta[c];
    float z = zp[(size_t)c << 12];
    v *= z / (1.f + expf(-z));
    out[(((size_t)(b * DIc + c)) << 12) + l] = (__bf16)v;
  }
}

// ---------------------------------------------------------------
extern "C" void kernel_launch(void* const* d_in, const int* in_sizes, int n_in,
                              void* d_out, int out_size, void* d_ws, size_t ws_size,
                              hipStream_t stream) {
  (void)in_sizes; (void)n_in; (void)out_size; (void)ws_size;
  const float* x        = (const float*)d_in[0];
  const float* conv1_w  = (const float*)d_in[1];
  const float* bn1_g    = (const float*)d_in[2];
  const float* bn1_b    = (const float*)d_in[3];
  const float* bn1_m    = (const float*)d_in[4];
  const float* bn1_v    = (const float*)d_in[5];
  const float* conv2_w  = (const float*)d_in[6];
  const float* bn2_g    = (const float*)d_in[7];
  const float* bn2_b    = (const float*)d_in[8];
  const float* bn2_m    = (const float*)d_in[9];
  const float* bn2_v    = (const float*)d_in[10];
  const float* ln_g     = (const float*)d_in[11];
  const float* ln_b     = (const float*)d_in[12];
  const float* in_pw    = (const float*)d_in[13];
  const float* in_pb    = (const float*)d_in[14];
  const float* dw_w     = (const float*)d_in[15];
  const float* dw_b     = (const float*)d_in[16];
  const float* x_pw     = (const float*)d_in[17];
  const float* dt_pw    = (const float*)d_in[18];
  const float* dt_pb    = (const float*)d_in[19];
  const float* A_log    = (const float*)d_in[20];
  const float* Ds       = (const float*)d_in[21];
  const float* on_g     = (const float*)d_in[22];
  const float* on_b     = (const float*)d_in[23];
  const float* out_pw   = (const float*)d_in[24];
  const float* out_pb   = (const float*)d_in[25];
  float* out = (float*)d_out;

  char* ws = (char*)d_ws;
  size_t off = 0;
  auto carve = [&](size_t bytes) -> void* {
    void* p = (void*)(ws + off);
    off = (off + bytes + 255) & ~(size_t)255;
    return p;
  };
  const int nX   = Bsz * CIN * Lpix;
  const int nH   = Bsz * COUT * Lpix;
  const int nXZ  = Bsz * 2 * DIc * Lpix;
  const int nXC  = Bsz * DIc * Lpix;
  const int nXD  = Bsz * Kdir * 40 * Lpix;

  __bf16* x_bf   = (__bf16*)carve((size_t)nX * 2);
  __bf16* w1_bf  = (__bf16*)carve((size_t)COUT * CIN * 9 * 2);
  __bf16* w2_bf  = (__bf16*)carve((size_t)COUT * COUT * 9 * 2);
  __bf16* wip_bf = (__bf16*)carve((size_t)2 * DIc * COUT * 2);
  __bf16* wxp_bf = (__bf16*)carve((size_t)Kdir * 40 * DIc * 2);
  __bf16* wop_bf = (__bf16*)carve((size_t)COUT * DIc * 2);
  float*  hcnn1  = (float*)carve((size_t)nH * 4);
  __bf16* h1_bf  = (__bf16*)carve((size_t)nH * 2);
  float*  hcnn   = (float*)carve((size_t)nH * 4);
  __bf16* xn_bf  = (__bf16*)carve((size_t)nH * 2);
  float*  xz     = (float*)carve((size_t)nXZ * 4);
  float*  xc     = (float*)carve((size_t)nXC * 4);
  __bf16* xc_bf  = (__bf16*)carve((size_t)nXC * 2);
  float*  xdbl   = (float*)carve((size_t)nXD * 4);
  float*  ybuf   = (float*)carve((size_t)nXC * 4);
  __bf16* yg_bf  = (__bf16*)carve((size_t)nXC * 2);

  auto cgrid = [](int n) { return dim3((unsigned)((n + 255) / 256)); };

  // stage bf16 operands (conv weights permuted to [co][tap][ci])
  cast_to_bf16<<<cgrid(nX), 256, 0, stream>>>(x, x_bf, nX);
  permute_conv_w_bf16<<<cgrid(COUT * CIN * 9), 256, 0, stream>>>(conv1_w, w1_bf, COUT, CIN);
  permute_conv_w_bf16<<<cgrid(COUT * COUT * 9), 256, 0, stream>>>(conv2_w, w2_bf, COUT, COUT);
  cast_to_bf16<<<cgrid(2 * DIc * COUT), 256, 0, stream>>>(in_pw, wip_bf, 2 * DIc * COUT);
  cast_to_bf16<<<cgrid(Kdir * 40 * DIc), 256, 0, stream>>>(x_pw, wxp_bf, Kdir * 40 * DIc);
  cast_to_bf16<<<cgrid(COUT * DIc), 256, 0, stream>>>(out_pw, wop_bf, COUT * DIc);

  const int pixTiles = Bsz * Lpix / 16;   // 2048
  dim3 blk(32, 4);

  // conv1 + bn + relu
  conv_bn_relu_wmma<<<dim3(pixTiles, 1), blk, 0, stream>>>(
      x_bf, w1_bf, bn1_g, bn1_b, bn1_m, bn1_v, hcnn1, CIN);
  cast_to_bf16<<<cgrid(nH), 256, 0, stream>>>(hcnn1, h1_bf, nH);

  // conv2 + bn + relu
  conv_bn_relu_wmma<<<dim3(pixTiles, 1), blk, 0, stream>>>(
      h1_bf, w2_bf, bn2_g, bn2_b, bn2_m, bn2_v, hcnn, COUT);

  // LN (channels-first) -> bf16
  ln_channels_bf16<<<cgrid(Bsz * Lpix), 256, 0, stream>>>(hcnn, ln_g, ln_b, xn_bf, COUT);

  // in_proj: M=512 (grid.y = 4)
  gemm_rowL_wmma<<<dim3(pixTiles, 4), blk, 0, stream>>>(
      wip_bf, xn_bf, in_pb, (const float*)nullptr, xz, 2 * DIc, COUT);

  // depthwise conv + SiLU
  dwconv_silu<<<cgrid(nXC), 256, 0, stream>>>(xz, dw_w, dw_b, xc, xc_bf);

  // x_proj per (b, direction)
  xproj_wmma<<<dim3(Bsz * Kdir * (Lpix / 16)), dim3(32, 3), 0, stream>>>(wxp_bf, xc_bf, xdbl);

  // selective scan (4 directions fused via atomic scatter)
  zero_f32<<<cgrid(nXC), 256, 0, stream>>>(ybuf, nXC);
  selective_scan<<<cgrid(Bsz * Kdir * DIc), 256, 0, stream>>>(
      xdbl, xc, dt_pw, dt_pb, A_log, Ds, ybuf);

  // out_norm LN * SiLU(z) -> bf16
  ln_gate_bf16<<<cgrid(Bsz * Lpix), 256, 0, stream>>>(ybuf, xz, on_g, on_b, yg_bf);

  // out_proj: M=128 + bias + residual
  gemm_rowL_wmma<<<dim3(pixTiles, 1), blk, 0, stream>>>(
      wop_bf, yg_bf, out_pb, hcnn, out, COUT, DIc);
}
